// VARS_D_43834436223172
// MI455X (gfx1250) — matrix operand order repaired and checked by reading, not compiled, not run on hardware
//
#include <hip/hip_runtime.h>
#include <hip/hip_bf16.h>

// ---------------------------------------------------------------------------
// Problem constants (from the reference)
// ---------------------------------------------------------------------------
#define BB 8
#define NN 4096
#define CC 768
#define HH 12
#define HD 64
#define MM 128          // rand feature dim
#define LAM 0.3f
#define NUM_STEP 5
#define KSPLIT 8        // K-dimension split for k_gp

typedef __bf16 bf16_t;
typedef __attribute__((ext_vector_type(16))) __bf16 v16bf;
typedef __attribute__((ext_vector_type(8)))  __bf16 v8bf;
typedef __attribute__((ext_vector_type(8)))  float  v8f;

// ---------------------------------------------------------------------------
// WMMA fragment helpers (CDNA5 wave32 layouts, CDNA5 ISA §7.12.2)
//
// A 16x32 bf16 (MxK): lanes 0-15 -> M=lane, elems[0..7]=K 0..7, [8..15]=K16..23
//                     lanes16-31 -> M=lane-16, elems[0..7]=K 8..15, [8..15]=K24..31
//   -> from a contiguous 32-element K-row: two b128 loads at +kb and +16+kb
// B 32x16 bf16 (KxN): lanes 0-15 -> N=lane, K 0..7,16..23; lanes16-31 -> K 8..15,24..31
//   -> from an N-major (transposed) 32-element K-column: two b128 loads
// C/D 16x16 f32: elem r at (M = r + 8*(lane>>4), N = lane&15)
// ---------------------------------------------------------------------------
__device__ __forceinline__ v16bf frag_a32(const bf16_t* row) {
  const int kb = ((threadIdx.x & 31) >> 4) << 3;
  v8bf lo = *(const v8bf*)(row + kb);
  v8bf hi = *(const v8bf*)(row + 16 + kb);
  v16bf a;
#pragma unroll
  for (int j = 0; j < 8; ++j) { a[j] = lo[j]; a[8 + j] = hi[j]; }
  return a;
}

__device__ __forceinline__ v16bf frag_b32(const bf16_t* col) {
  const int kb = ((threadIdx.x & 31) >> 4) << 3;
  v8bf lo = *(const v8bf*)(col + kb);
  v8bf hi = *(const v8bf*)(col + 16 + kb);
  v16bf b;
#pragma unroll
  for (int j = 0; j < 8; ++j) { b[j] = lo[j]; b[8 + j] = hi[j]; }
  return b;
}

__device__ __forceinline__ v8f wmma_bf16(v16bf a, v16bf b, v8f c) {
  return __builtin_amdgcn_wmma_f32_16x16x32_bf16(false, a, false, b,
                                                 (short)0, c, false, false);
}

__device__ __forceinline__ float softt(float z, float t) {
  float az = fabsf(z) - t;
  return az > 0.f ? copysignf(az, z) : 0.f;
}

// Async DMA: global -> LDS, 16 bytes per enabled lane (CDNA5, ASYNCcnt).
// ldsaddr = LDS byte offset (low 32 bits of generic pointer, ISA 10.2).
__device__ __forceinline__ void async_g2l_b128(unsigned ldsaddr, const void* g) {
  asm volatile("global_load_async_to_lds_b128 %0, %1, off"
               :: "v"(ldsaddr), "v"(g) : "memory");
}
__device__ __forceinline__ void wait_async0() {
  asm volatile("s_wait_asynccnt 0x0" ::: "memory");
}

// ---------------------------------------------------------------------------
// Kernel 0: zero an fp32 region (colss, invn, G, P accumulators)
// ---------------------------------------------------------------------------
__global__ void k_zero(float* __restrict__ p, int n) {
  int i = blockIdx.x * blockDim.x + threadIdx.x;
  if (i < n) p[i] = 0.f;
}

// ---------------------------------------------------------------------------
// Kernel 1: qkv = x @ Wqkv  ->  qs (scaled by hd^-0.25, bf16 [B,H,N,64])
//                               v  (bf16 [B,H,N,64])
// grid (12, 2048), block 256 (8 waves). Block tile: 16 rows x 128 cols.
// Per-thread loop-invariant staging pointers; fragment reads are ds b128.
// ---------------------------------------------------------------------------
__global__ void k_qkv(const float* __restrict__ x, const float* __restrict__ Wqkv,
                      bf16_t* __restrict__ qs, bf16_t* __restrict__ v) {
  __shared__ __align__(16) bf16_t lA[16 * 32];     // A tile (row-major, ld 32)
  __shared__ __align__(16) bf16_t lBT[128 * 32];   // W chunk transposed (col-major)
  const int wave = threadIdx.x >> 5;
  const int lane = threadIdx.x & 31;
  const int tid  = threadIdx.x;
  const int cb = blockIdx.x * 128;   // column base over 1536
  const int rt = blockIdx.y;         // 0..2047 row tiles over 32768
  const float* xrow = x + (size_t)rt * 16 * CC;

  // loop-invariant staging decomposition:
  //  A: thread -> row r=tid>>4, k-pair k=(tid&15)*2 ; moves by +kc each chunk
  const float*  xa  = xrow + (size_t)(tid >> 4) * CC + (tid & 15) * 2;
  bf16_t*       lAa = lA + tid * 2;
  //  B: idx = tid + 256*t -> k = (tid>>7) + 2t, c = tid&127
  const float*  wb  = Wqkv + (size_t)(tid >> 7) * (2 * CC) + cb + (tid & 127);
  bf16_t*       lBb = lBT + (tid & 127) * 32 + (tid >> 7);

  v8f acc = {};
#pragma unroll 1
  for (int kc = 0; kc < CC; kc += 32) {
    __syncthreads();
    {
      float2 xv = *(const float2*)(xa + kc);
      lAa[0] = (bf16_t)xv.x;
      lAa[1] = (bf16_t)xv.y;
    }
    {
      const float* p = wb + (size_t)kc * (2 * CC);
#pragma unroll
      for (int t = 0; t < 16; ++t)
        lBb[2 * t] = (bf16_t)p[(size_t)t * (4 * CC)];   // k += 2 per t
    }
    if (kc + 32 < CC) {
      __builtin_prefetch(xa + kc + 32, 0, 0);
      __builtin_prefetch(wb + (size_t)(kc + 32) * (2 * CC), 0, 0);
    }
    __syncthreads();
    v16bf a = frag_a32(lA + (lane & 15) * 32);
    v16bf b = frag_b32(lBT + (wave * 16 + (lane & 15)) * 32);
    acc = wmma_bf16(a, b, acc);
  }
  const int nn = lane & 15;
  const int hi = lane >> 4;
  const int col = cb + wave * 16 + nn;
  const bool isq = (col < CC);
  const int col2 = isq ? col : (col - CC);
  const int h = col2 >> 6;
  const int d = col2 & 63;
  const float qscale = 0.35355339059327373f;   // hd^-0.25
#pragma unroll
  for (int r = 0; r < 8; ++r) {
    const int g  = rt * 16 + r + (hi << 3);    // global row in [0,32768)
    const int b_ = g >> 12;
    const int ns = g & 4095;
    const size_t idx = (((size_t)(b_ * HH + h)) * NN + ns) * HD + d;
    float val = acc[r];
    if (isq) qs[idx] = (bf16_t)(val * qscale);
    else     v[idx]  = (bf16_t)val;
  }
}

// ---------------------------------------------------------------------------
// Kernel 2: rf = exp(qs @ R[h] - 0.5*||qs||^2) / sqrt(m)  (bf16 [B,H,N,128])
//           + atomic accumulation of per-column sum of squares (for L2 norm)
// grid (256 n-tiles, 96 bh), block 256; wave w = m-tile w.
// R[h] staged transposed in LDS (16 KB) once per block.
// ---------------------------------------------------------------------------
__global__ void k_rf(const bf16_t* __restrict__ qs, const float* __restrict__ R,
                     bf16_t* __restrict__ rf, float* __restrict__ colss) {
  __shared__ __align__(16) bf16_t lRT[128 * 64];   // R[h] transposed [m][k]
  const int mt = threadIdx.x >> 5;  // 0..7
  const int lane = threadIdx.x & 31;
  const int tid  = threadIdx.x;
  const int nt = blockIdx.x;        // 0..255
  const int bh = blockIdx.y;        // 0..95
  const int h = bh % HH;

  // stage R[h] [64 k x 128 m] fp32 -> lRT[m*64 + k] bf16
  // idx = tid + 256*t -> k = (tid>>7) + 2t, m = tid&127
  {
    const float* rs = R + (size_t)h * HD * MM + (size_t)(tid >> 7) * MM + (tid & 127);
    bf16_t* ld = lRT + (tid & 127) * 64 + (tid >> 7);
#pragma unroll
    for (int t = 0; t < 32; ++t)
      ld[2 * t] = (bf16_t)rs[(size_t)t * (2 * MM)];
  }
  __syncthreads();

  const bf16_t* qbase = qs + (((size_t)bh * NN) + nt * 16) * HD;
  const bf16_t* qrow = qbase + (size_t)(lane & 15) * HD;
  const bf16_t* rcol = lRT + (size_t)(mt * 16 + (lane & 15)) * 64;

  v8f acc = {};
  acc = wmma_bf16(frag_a32(qrow),      frag_b32(rcol),      acc);
  acc = wmma_bf16(frag_a32(qrow + 32), frag_b32(rcol + 32), acc);

  // per-row sum of squares of (already scaled) qs; rows 0..7 in lanes 0..7,
  // rows 8..15 in lanes 16..23  (vectorized b128 loads)
  const int rowl = (lane & 7) | ((lane >> 4) << 3);
  const bf16_t* rq = qbase + (size_t)rowl * HD;
  float ss = 0.f;
#pragma unroll
  for (int c = 0; c < 8; ++c) {
    v8bf qv = *(const v8bf*)(rq + c * 8);
#pragma unroll
    for (int j = 0; j < 8; ++j) { float t = (float)qv[j]; ss += t * t; }
  }

  const float rsm = 0.08838834764831845f;  // 1/sqrt(128)
  const int nn = lane & 15;
  const int hi = lane >> 4;
  bf16_t* rfb = rf + (((size_t)bh * NN) + nt * 16) * MM + mt * 16 + nn;
  float cp = 0.f;
#pragma unroll
  for (int r = 0; r < 8; ++r) {
    float ssr  = __shfl(ss, r + (lane & 16), 32);
    float valf = expf(acc[r] - 0.5f * ssr) * rsm;
    rfb[(size_t)(r + (hi << 3)) * MM] = (bf16_t)valf;
    cp += valf * valf;
  }
  cp += __shfl_xor(cp, 16, 32);              // fold both row halves of column
  if (lane < 16) atomicAdd(&colss[bh * MM + mt * 16 + lane], cp);
}

// ---------------------------------------------------------------------------
// Kernel 3: invnorm = 1 / max(sqrt(colss), 1e-12)
// ---------------------------------------------------------------------------
__global__ void k_inv(const float* __restrict__ ss, float* __restrict__ inv, int n) {
  int i = blockIdx.x * blockDim.x + threadIdx.x;
  if (i < n) inv[i] = 1.f / fmaxf(sqrtf(ss[i]), 1e-12f);
}

// ---------------------------------------------------------------------------
// Kernel 4: G = rf^T rf [128x128], P = rf^T v [128x64]  per (b,h)
// grid (KSPLIT=8, 96), block 256 (8 waves). Wave w owns row strip mi=w:
//   8 G tiles + 4 P tiles = 12 WMMAs per K-chunk sharing ONE A fragment.
// rf/v chunks staged transposed in LDS; accumulated to global via atomicAdd.
// ---------------------------------------------------------------------------
__global__ void k_gp(const bf16_t* __restrict__ rf, const bf16_t* __restrict__ v,
                     float* __restrict__ G, float* __restrict__ P) {
  __shared__ __align__(16) bf16_t lRT[128 * 32];   // rf chunk transposed [m][k]
  __shared__ __align__(16) bf16_t lVT[64 * 32];    // v  chunk transposed [d][k]
  const int wave = threadIdx.x >> 5;   // = mi (0..7)
  const int lane = threadIdx.x & 31;
  const int tid  = threadIdx.x;
  const int bh = blockIdx.y;
  const int k0 = blockIdx.x * (NN / KSPLIT);
  const bf16_t* rfb = rf + (size_t)bh * NN * MM;
  const bf16_t* vb  = v  + (size_t)bh * NN * HD;

  // staging decomposition (loop-invariant per thread):
  //  rf: idx = tid + 256*t -> k = (tid>>7) + 2t, m = tid&127
  const bf16_t* rsrc = rfb + (size_t)(tid >> 7) * MM + (tid & 127);
  bf16_t*       rdst = lRT + (tid & 127) * 32 + (tid >> 7);
  //  v:  idx = tid + 256*t -> k = (tid>>6) + 4t, d = tid&63
  const bf16_t* vsrc = vb + (size_t)(tid >> 6) * HD + (tid & 63);
  bf16_t*       vdst = lVT + (tid & 63) * 32 + (tid >> 6);

  v8f accG[8] = {};
  v8f accP[4] = {};

#pragma unroll 1
  for (int kc = k0; kc < k0 + NN / KSPLIT; kc += 32) {
    __syncthreads();
    {
      const bf16_t* p = rsrc + (size_t)kc * MM;
#pragma unroll
      for (int t = 0; t < 16; ++t) rdst[2 * t] = p[(size_t)t * (2 * MM)];
    }
    {
      const bf16_t* p = vsrc + (size_t)kc * HD;
#pragma unroll
      for (int t = 0; t < 8; ++t) vdst[4 * t] = p[(size_t)t * (4 * HD)];
    }
    if (kc + 32 < k0 + NN / KSPLIT) {
      __builtin_prefetch(rsrc + (size_t)(kc + 32) * MM, 0, 0);
      __builtin_prefetch(vsrc + (size_t)(kc + 32) * HD, 0, 0);
    }
    __syncthreads();
    v16bf a = frag_a32(lRT + (size_t)(wave * 16 + (lane & 15)) * 32);
#pragma unroll
    for (int mj = 0; mj < 8; ++mj) {
      v16bf b = frag_b32(lRT + (size_t)(mj * 16 + (lane & 15)) * 32);
      accG[mj] = wmma_bf16(a, b, accG[mj]);
    }
#pragma unroll
    for (int dj = 0; dj < 4; ++dj) {
      v16bf b = frag_b32(lVT + (size_t)(dj * 16 + (lane & 15)) * 32);
      accP[dj] = wmma_bf16(a, b, accP[dj]);
    }
  }

  const int nn = lane & 15, hi = lane >> 4;
  float* Gb = G + (size_t)bh * MM * MM + (size_t)(wave * 16) * MM;
  float* Pb = P + (size_t)bh * MM * HD + (size_t)(wave * 16) * HD;
#pragma unroll
  for (int mj = 0; mj < 8; ++mj)
#pragma unroll
    for (int r = 0; r < 8; ++r)
      atomicAdd(&Gb[(size_t)(r + (hi << 3)) * MM + mj * 16 + nn], accG[mj][r]);
#pragma unroll
  for (int dj = 0; dj < 4; ++dj)
#pragma unroll
    for (int r = 0; r < 8; ++r)
      atomicAdd(&Pb[(size_t)(r + (hi << 3)) * HD + dj * 16 + nn], accP[dj][r]);
}

// ---------------------------------------------------------------------------
// Kernel 5: per-(b,h) ISTA in LDS.
//   kk = D G D, kv = D P, L = max row-L1(kk)+1, 5 soft-threshold steps,
//   emit sd = D*s (bf16) so the final GEMM is rf @ sd  (== qn @ s).
// grid 96, block 256, dynamic LDS ~132 KB (WGP has 320 KB).
// ---------------------------------------------------------------------------
__global__ void k_ista(const float* __restrict__ G, const float* __restrict__ P,
                       const float* __restrict__ invn, bf16_t* __restrict__ sd) {
  extern __shared__ float smem[];
  float* kk     = smem;                  // 128*128
  float* kv     = kk + MM * MM;          // 128*64
  float* s      = kv + MM * HD;          // 128*64
  float* invs   = s + MM * HD;           // 128
  float* rowsum = invs + MM;             // 128
  float* Lp     = rowsum + MM;           // 1

  const int bh  = blockIdx.x;
  const int tid = threadIdx.x;           // 256 threads
  const float* Gb = G + (size_t)bh * MM * MM;
  const float* Pb = P + (size_t)bh * MM * HD;

  if (tid < MM) invs[tid] = invn[bh * MM + tid];
  __syncthreads();

  for (int idx = tid; idx < MM * MM; idx += 256) {
    int i = idx >> 7, j = idx & 127;
    kk[idx] = Gb[idx] * invs[i] * invs[j];
  }
  for (int idx = tid; idx < MM * HD; idx += 256) {
    int i = idx >> 6;
    kv[idx] = Pb[idx] * invs[i];
  }
  __syncthreads();

  if (tid < MM) {
    float rs = 0.f;
    for (int j = 0; j < MM; ++j) rs += fabsf(kk[tid * MM + j]);
    rowsum[tid] = rs;
  }
  __syncthreads();
  if (tid == 0) {
    float mx = 0.f;
    for (int j = 0; j < MM; ++j) mx = fmaxf(mx, rowsum[j]);
    Lp[0] = mx + 1.f;
  }
  __syncthreads();
  const float L = Lp[0];
  const float invL = 1.f / L;
  const float thr  = LAM * invL;

  for (int idx = tid; idx < MM * HD; idx += 256) s[idx] = softt(kv[idx], LAM);
  __syncthreads();

  float tmp[32];
  for (int step = 0; step < NUM_STEP; ++step) {
    int c = 0;
    for (int idx = tid; idx < MM * HD; idx += 256, ++c) {
      int i = idx >> 6, d = idx & 63;
      float acc = 0.f;
      for (int k2 = 0; k2 < MM; ++k2) acc += kk[i * MM + k2] * s[k2 * HD + d];
      tmp[c] = softt(s[idx] - (acc - kv[idx]) * invL, thr);
    }
    __syncthreads();
    c = 0;
    for (int idx = tid; idx < MM * HD; idx += 256, ++c) s[idx] = tmp[c];
    __syncthreads();
  }

  bf16_t* sdb = sd + (size_t)bh * MM * HD;
  for (int idx = tid; idx < MM * HD; idx += 256) {
    int i = idx >> 6;
    sdb[idx] = (bf16_t)(s[idx] * invs[i]);
  }
}

// ---------------------------------------------------------------------------
// Kernel 6: y[b, n, h*64 + d] = (rf @ sd)  bf16  -- per (b,h) [4096x128]@[128x64]
// grid (128, 96), block 256. sd^T staged in LDS (16 KB); rf A frags are
// contiguous b128 global loads.
// ---------------------------------------------------------------------------
__global__ void k_out(const bf16_t* __restrict__ rf, const bf16_t* __restrict__ sd,
                      bf16_t* __restrict__ y) {
  __shared__ __align__(16) bf16_t lST[64 * 128];   // sd transposed [d][m]
  const int wave = threadIdx.x >> 5;
  const int lane = threadIdx.x & 31;
  const int tid  = threadIdx.x;
  const int bh = blockIdx.y;
  // stage sd [128 m x 64 d] -> lST[d*128 + m]
  // idx = tid + 256*t -> m = (tid>>6) + 4t, d = tid&63
  {
    const bf16_t* p = sd + (size_t)bh * MM * HD + tid;
    bf16_t* ld = lST + (tid & 63) * 128 + (tid >> 6);
#pragma unroll
    for (int t = 0; t < 32; ++t) ld[4 * t] = p[256 * t];
  }
  __syncthreads();

  const int t = blockIdx.x * 8 + wave;   // 0..1023
  const int nt = t >> 2, dt = t & 3;
  const int b_ = bh / HH, h = bh % HH;
  const bf16_t* arow =
      rf + ((size_t)bh * NN + nt * 16 + (lane & 15)) * MM;
  const bf16_t* bcol = lST + (size_t)(dt * 16 + (lane & 15)) * 128;
  v8f acc = {};
#pragma unroll
  for (int c = 0; c < 4; ++c)
    acc = wmma_bf16(frag_a32(arow + c * 32), frag_b32(bcol + c * 32), acc);

  const int nn = lane & 15, hi = lane >> 4;
  const int col = h * HD + dt * 16 + nn;
#pragma unroll
  for (int r = 0; r < 8; ++r) {
    const int row = b_ * NN + nt * 16 + r + (hi << 3);
    y[(size_t)row * CC + col] = (bf16_t)acc[r];
  }
}

// ---------------------------------------------------------------------------
// Kernel 7: out = y @ Wproj + bproj   (fp32 out [32768x768])
// grid (6, 2048), block 256. A tile is DMA'd global->LDS with the CDNA5
// async-tensor path (GLOBAL_LOAD_ASYNC_TO_LDS_B128 + s_wait_asynccnt).
// ---------------------------------------------------------------------------
__global__ void k_proj(const bf16_t* __restrict__ y, const float* __restrict__ Wp,
                       const float* __restrict__ bias, float* __restrict__ out) {
  __shared__ __align__(16) bf16_t lA[16 * 32];
  __shared__ __align__(16) bf16_t lBT[128 * 32];
  const int wave = threadIdx.x >> 5;
  const int lane = threadIdx.x & 31;
  const int tid  = threadIdx.x;
  const int cb = blockIdx.x * 128;   // 0..640
  const int rt = blockIdx.y;         // 0..2047
  const bf16_t* yrow = y + (size_t)rt * 16 * CC;

  // async A staging: 16 rows x 64 B = 64 x b128 transfers (waves 0-1)
  // lane tid<64: row = tid>>2, 16-byte segment = tid&3
  const unsigned ldsA = ((unsigned)(uintptr_t)(&lA[0])) + tid * 16;
  const bf16_t*  ga   = yrow + (size_t)(tid >> 2) * CC + (tid & 3) * 8;
  // B staging: idx = tid + 256*t -> k = (tid>>7) + 2t, c = tid&127
  const float*   wb   = Wp + (size_t)(tid >> 7) * CC + cb + (tid & 127);
  bf16_t*        lBb  = lBT + (tid & 127) * 32 + (tid >> 7);

  v8f acc = {};
#pragma unroll 1
  for (int kc = 0; kc < CC; kc += 32) {
    __syncthreads();
    if (tid < 64) async_g2l_b128(ldsA, ga + kc);   // DMA 16B/lane -> lA
    {
      const float* p = wb + (size_t)kc * CC;
#pragma unroll
      for (int t = 0; t < 16; ++t)
        lBb[2 * t] = (bf16_t)p[(size_t)t * (2 * CC)];
    }
    if (kc + 32 < CC) __builtin_prefetch(wb + (size_t)(kc + 32) * CC, 0, 0);
    wait_async0();                                  // ASYNCcnt==0 before barrier
    __syncthreads();
    v16bf a = frag_a32(lA + (lane & 15) * 32);
    v16bf b = frag_b32(lBT + (wave * 16 + (lane & 15)) * 32);
    acc = wmma_bf16(a, b, acc);
  }
  const int nn = lane & 15, hi = lane >> 4;
  const int c0 = cb + wave * 16;
  const float bcol = bias[c0 + nn];
#pragma unroll
  for (int r = 0; r < 8; ++r) {
    const int g = rt * 16 + r + (hi << 3);
    out[(size_t)g * CC + c0 + nn] = acc[r] + bcol;
  }
}

// ---------------------------------------------------------------------------
// Host launcher
// ---------------------------------------------------------------------------
extern "C" void kernel_launch(void* const* d_in, const int* in_sizes, int n_in,
                              void* d_out, int out_size, void* d_ws, size_t ws_size,
                              hipStream_t stream) {
  const float* x     = (const float*)d_in[0];
  const float* Wqkv  = (const float*)d_in[1];
  const float* Wproj = (const float*)d_in[2];
  const float* bproj = (const float*)d_in[3];
  const float* R     = (const float*)d_in[4];
  float* out = (float*)d_out;

  char* ws = (char*)d_ws;
  size_t off = 0;
  auto alloc = [&](size_t bytes) -> void* {
    void* p = ws + off;
    off += (bytes + 255) & ~(size_t)255;
    return p;
  };

  const size_t nBHN = (size_t)BB * HH * NN;
  bf16_t* qs    = (bf16_t*)alloc(nBHN * HD * sizeof(bf16_t));     // 50.3 MB
  bf16_t* v     = (bf16_t*)alloc(nBHN * HD * sizeof(bf16_t));     // 50.3 MB
  bf16_t* rf    = (bf16_t*)alloc(nBHN * MM * sizeof(bf16_t));     // 100.7 MB
  float*  colss = (float*) alloc((size_t)BB * HH * MM * sizeof(float));
  float*  invn  = (float*) alloc((size_t)BB * HH * MM * sizeof(float));
  float*  G     = (float*) alloc((size_t)BB * HH * MM * MM * sizeof(float));
  float*  P     = (float*) alloc((size_t)BB * HH * MM * HD * sizeof(float));
  bf16_t* sd    = (bf16_t*)alloc((size_t)BB * HH * MM * HD * sizeof(bf16_t));
  bf16_t* y     = (bf16_t*)alloc((size_t)BB * NN * CC * sizeof(bf16_t));

  // zero colss + invn + G + P in one shot (contiguous, 256B-aligned sizes)
  const int nZero = BB * HH * MM             // colss
                  + BB * HH * MM             // invn (overwritten later, harmless)
                  + BB * HH * MM * MM        // G
                  + BB * HH * MM * HD;       // P
  k_zero<<<dim3((nZero + 255) / 256), dim3(256), 0, stream>>>(colss, nZero);

  // qkv GEMM: block tile 16 rows x 128 cols
  k_qkv<<<dim3(12, 2048), dim3(256), 0, stream>>>(x, Wqkv, qs, v);

  // random-feature map
  k_rf<<<dim3(256, 96), dim3(256), 0, stream>>>(qs, R, rf, colss);

  const int nCol = BB * HH * MM;             // 12288
  k_inv<<<dim3((nCol + 255) / 256), dim3(256), 0, stream>>>(colss, invn, nCol);

  // G = rf^T rf, P = rf^T v  (K split 8 ways, atomic reduction)
  k_gp<<<dim3(KSPLIT, 96), dim3(256), 0, stream>>>(rf, v, G, P);

  // ISTA (dynamic LDS: kk + kv + s + invs + rowsum + L)
  const size_t smemBytes =
      (size_t)(MM * MM + MM * HD + MM * HD + MM + MM + 1) * sizeof(float);
  k_ista<<<dim3(96), dim3(256), smemBytes, stream>>>(G, P, invn, sd);

  // attention output: rf @ (D s)
  k_out<<<dim3(128, 96), dim3(256), 0, stream>>>(rf, sd, y);

  // final projection + bias
  k_proj<<<dim3(6, 2048), dim3(256), 0, stream>>>(y, Wproj, bproj, out);
}
